// Local_EncoderLayer_15367392985812
// MI455X (gfx1250) — compile-verified, matrix operand order checked by reading, not run on hardware
//
#include <hip/hip_runtime.h>
#include <math.h>

// ---------------------------------------------------------------------------
// MI455X (gfx1250) transformer encoder layer.
// All GEMMs lowered to v_wmma_f32_16x16x32_bf16 (wave32, 16x16 tiles).
// f32 inputs are cast once to bf16; accumulation stays f32.
// Big GEMMs: 128x128 block tile, 8 waves (4x2), 32x64 wave tile (2x4 frags).
// Double-buffered LDS pipeline; A (and transposed-B) tiles staged with
// GLOBAL_LOAD_ASYNC_TO_LDS_B128 (ASYNCcnt) when the builtin is available.
// ---------------------------------------------------------------------------

typedef unsigned short u16;
typedef unsigned int   u32;
typedef __attribute__((ext_vector_type(16))) __bf16 v16bf;
typedef __attribute__((ext_vector_type(8)))  float  v8f;

#define LN_EPS 1e-6f
#define BK 32
#define LDP 40   // LDS row pitch (u16): 80B = 16B-aligned rows, 20-bank stride

#if defined(__AMDGCN__) && __has_builtin(__builtin_amdgcn_global_load_async_to_lds_b128)
#define ASYNC_LDS 1
typedef int v4i __attribute__((vector_size(16)));
typedef __attribute__((address_space(1))) v4i gv4i;   // global-AS int4
typedef __attribute__((address_space(3))) v4i lv4i;   // LDS-AS int4
#else
#define ASYNC_LDS 0
#endif

template <int N>
__device__ __forceinline__ void wait_async_n() {
#if ASYNC_LDS
#if __has_builtin(__builtin_amdgcn_s_wait_asynccnt)
  __builtin_amdgcn_s_wait_asynccnt((unsigned short)N);
#else
  asm volatile("s_wait_asynccnt %0" ::"i"(N) : "memory");
#endif
#endif
}

__device__ __forceinline__ u16 f2bf(float f) {
  u32 u = __float_as_uint(f);
  u += 0x7fffu + ((u >> 16) & 1u);   // round-to-nearest-even
  return (u16)(u >> 16);
}

// ------------------------------- casts -------------------------------------
__global__ __launch_bounds__(256) void cast_f32_bf16_k(
    const float* __restrict__ x, u16* __restrict__ y, size_t n) {
  size_t i = (size_t)blockIdx.x * blockDim.x + threadIdx.x;
  if (i < n) y[i] = f2bf(x[i]);
}

// w [H][D][dk] f32  ->  o [D][H*dk] bf16   (head-major weights to GEMM-B layout)
__global__ __launch_bounds__(256) void cast_head_w_k(
    const float* __restrict__ w, u16* __restrict__ o, int H, int D, int dk) {
  size_t n = (size_t)H * D * dk;
  size_t i = (size_t)blockIdx.x * blockDim.x + threadIdx.x;
  if (i >= n) return;
  int h = (int)(i / ((size_t)D * dk));
  size_t rem = i - (size_t)h * D * dk;
  int k = (int)(rem / dk), j = (int)(rem % dk);
  o[(size_t)k * ((size_t)H * dk) + (size_t)h * dk + j] = f2bf(w[i]);
}

// ----------------------- strided-batch WMMA GEMM ---------------------------
// C[M,N] = alpha * A[M,K] * B(K,N) (+bias) (+relu); A,B bf16; C f32 and/or bf16.
// Template: BMT x BNT block tile, WR x WC wave grid (wave32), K-step 32.
template <int BMT, int BNT, int WR, int WC, bool TRANSB, bool RELU>
__global__ __launch_bounds__(WR * WC * 32) void gemm_bf16_wmma(
    const u16* __restrict__ A, long lda, long sAb, long sAh,
    const u16* __restrict__ B, long ldb, long sBb, long sBh,
    float* __restrict__ Cf, u16* __restrict__ Cb, long ldc, long sCb, long sCh,
    const float* __restrict__ bias, float alpha, int M, int N, int K, int heads) {
  constexpr int NT  = WR * WC * 32;          // threads per block
  constexpr int FM  = BMT / (WR * 16);       // 16x16 frags per wave (M)
  constexpr int FN  = BNT / (WC * 16);       // 16x16 frags per wave (N)
  constexpr int AIT = (BMT * (BK / 8)) / NT; // A-tile chunks per thread
  constexpr int BIT = (BNT * (BK / 8)) / NT; // B-tile chunks per thread (TRANSB)
  constexpr int CPR = BNT / 8;               // chunks per k-row (!TRANSB)
  constexpr int BIT2 = (BK * CPR) / NT;      // B-tile chunks per thread (!TRANSB)
  constexpr int APT = ASYNC_LDS ? (TRANSB ? AIT + BIT : AIT) : 0; // asyncs/tile

  __shared__ u16 As[2][BMT][LDP];
  __shared__ u16 Bs[2][BNT][LDP];

  const int tid  = threadIdx.x;
  const int lane = tid & 31;
  const int wave = tid >> 5;
  const int wm = (wave / WC) * (FM * 16);
  const int wn = (wave % WC) * (FN * 16);

  const int z  = blockIdx.z;
  const int hh = z % heads;
  const int bb = z / heads;
  const u16* Ab = A + (size_t)bb * sAb + (size_t)hh * sAh;
  const u16* Bb = B + (size_t)bb * sBb + (size_t)hh * sBh;

  const int m0 = blockIdx.y * BMT;
  const int n0 = blockIdx.x * BNT;

  u32 regA[AIT][4];
  u32 regB[TRANSB ? BIT : BIT2][4];

  // ---- tile staging helpers ----
  auto issueA = [&](int kt, int buf) {
#pragma unroll
    for (int it = 0; it < AIT; ++it) {
      int c = tid + it * NT;
      int r = c >> 2, kc = (c & 3) << 3;
      const u16* src = Ab + (size_t)(m0 + r) * lda + kt + kc;
#if ASYNC_LDS
      __builtin_amdgcn_global_load_async_to_lds_b128(
          (gv4i*)src, (lv4i*)&As[buf][r][kc], 0, 0);
#else
      const u32* s = (const u32*)src;
      regA[it][0] = s[0]; regA[it][1] = s[1]; regA[it][2] = s[2]; regA[it][3] = s[3];
#endif
      (void)buf;
    }
  };
  auto storeA = [&](int buf) {
#if !ASYNC_LDS
#pragma unroll
    for (int it = 0; it < AIT; ++it) {
      int c = tid + it * NT;
      int r = c >> 2, kc = (c & 3) << 3;
      u32* d = (u32*)&As[buf][r][kc];
      d[0] = regA[it][0]; d[1] = regA[it][1]; d[2] = regA[it][2]; d[3] = regA[it][3];
    }
#else
    (void)buf;
#endif
  };
  auto issueB = [&](int kt, int buf) {
    if (TRANSB) {   // B given as [N,K] row-major -> contiguous along K
#pragma unroll
      for (int it = 0; it < BIT; ++it) {
        int c = tid + it * NT;
        int r = c >> 2, kc = (c & 3) << 3;
        const u16* src = Bb + (size_t)(n0 + r) * ldb + kt + kc;
#if ASYNC_LDS
        __builtin_amdgcn_global_load_async_to_lds_b128(
            (gv4i*)src, (lv4i*)&Bs[buf][r][kc], 0, 0);
#else
        const u32* s = (const u32*)src;
        regB[it][0] = s[0]; regB[it][1] = s[1]; regB[it][2] = s[2]; regB[it][3] = s[3];
#endif
      }
    } else {        // B given as [K,N] row-major -> stage regs, transpose on store
#pragma unroll
      for (int it = 0; it < BIT2; ++it) {
        int c = tid + it * NT;
        int k = c / CPR, nc = (c % CPR) << 3;
        const u32* s = (const u32*)(Bb + (size_t)(kt + k) * ldb + n0 + nc);
        regB[it][0] = s[0]; regB[it][1] = s[1]; regB[it][2] = s[2]; regB[it][3] = s[3];
      }
    }
    (void)buf;
  };
  auto storeB = [&](int buf) {
    if (TRANSB) {
#if !ASYNC_LDS
#pragma unroll
      for (int it = 0; it < BIT; ++it) {
        int c = tid + it * NT;
        int r = c >> 2, kc = (c & 3) << 3;
        u32* d = (u32*)&Bs[buf][r][kc];
        d[0] = regB[it][0]; d[1] = regB[it][1]; d[2] = regB[it][2]; d[3] = regB[it][3];
      }
#endif
    } else {
#pragma unroll
      for (int it = 0; it < BIT2; ++it) {
        int c = tid + it * NT;
        int k = c / CPR, nc = (c % CPR) << 3;
#pragma unroll
        for (int j = 0; j < 4; ++j) {
          u32 w = regB[it][j];
          Bs[buf][nc + 2 * j][k]     = (u16)(w & 0xffffu);
          Bs[buf][nc + 2 * j + 1][k] = (u16)(w >> 16);
        }
      }
    }
    (void)buf;
  };

  v8f acc[FM][FN] = {};

  // ---- double-buffered pipeline ----
  issueA(0, 0);
  issueB(0, 0);
  int p = 0;
  for (int kt = 0; kt < K; kt += BK) {
    const bool more = (kt + BK < K);
    __syncthreads();           // everyone done reading tile kt-1
    storeA(p); storeB(p);      // reg-staged data -> LDS (no-op on async paths)
    if (more) {                // overlap next tile's loads with this compute
      issueA(kt + BK, p ^ 1);
      issueB(kt + BK, p ^ 1);
      __builtin_prefetch(Ab + (size_t)(m0 + (tid % BMT)) * lda + kt + BK, 0, 1);
    }
    if (more) wait_async_n<APT>(); else wait_async_n<0>();
    __syncthreads();           // tile kt fully in LDS for all waves

    // fragment assembly per ISA 16-bit A-layout (K 0-7/16-23 vs 8-15/24-31 by lane[4])
    union Frag { v16bf v; u32 u[8]; };
    Frag a[FM], bfr[FN];
    const int ksel = (lane >> 4) << 3;
    const int rr = lane & 15;
#pragma unroll
    for (int f = 0; f < FM; ++f) {
      int ar = wm + f * 16 + rr;
#pragma unroll
      for (int i = 0; i < 4; ++i) {
        a[f].u[i]     = *(const u32*)&As[p][ar][ksel + 2 * i];
        a[f].u[4 + i] = *(const u32*)&As[p][ar][16 + ksel + 2 * i];
      }
    }
#pragma unroll
    for (int f = 0; f < FN; ++f) {
      int br = wn + f * 16 + rr;
#pragma unroll
      for (int i = 0; i < 4; ++i) {
        bfr[f].u[i]     = *(const u32*)&Bs[p][br][ksel + 2 * i];
        bfr[f].u[4 + i] = *(const u32*)&Bs[p][br][16 + ksel + 2 * i];
      }
    }
#pragma unroll
    for (int i = 0; i < FM; ++i)
#pragma unroll
      for (int j = 0; j < FN; ++j)
        acc[i][j] = __builtin_amdgcn_wmma_f32_16x16x32_bf16(
            false, a[i].v, false, bfr[j].v, (short)0, acc[i][j], false, false);
    p ^= 1;
  }

  // epilogue: C/D layout = 8 VGPRs, M = v + 8*lane[4], N = lane[3:0]
  float* Cfb = Cf ? Cf + (size_t)bb * sCb + (size_t)hh * sCh : nullptr;
  u16*  Cbb = Cb ? Cb + (size_t)bb * sCb + (size_t)hh * sCh : nullptr;
  const int col16 = lane & 15;
  const int radd  = (lane >> 4) << 3;
#pragma unroll
  for (int i = 0; i < FM; ++i)
#pragma unroll
    for (int j = 0; j < FN; ++j) {
      int colg = n0 + wn + j * 16 + col16;
      float bv = bias ? bias[colg] : 0.0f;
#pragma unroll
      for (int v = 0; v < 8; ++v) {
        int rowg = m0 + wm + i * 16 + radd + v;
        float val = acc[i][j][v] * alpha + bv;
        if (RELU) val = fmaxf(val, 0.0f);
        size_t off = (size_t)rowg * ldc + colg;
        if (Cfb) Cfb[off] = val;
        if (Cbb) Cbb[off] = f2bf(val);
      }
    }
}

// ------------------------- masked softmax (row) ----------------------------
// S: f32 [rows][n]; writes probs f32 in-place (optional) + bf16 copy.
__global__ __launch_bounds__(256) void softmax_mask_k(
    float* __restrict__ S, const unsigned char* __restrict__ mask,
    u16* __restrict__ P, int n, int slen, int heads, int storeF) {
  __shared__ float red[256];
  const int r = blockIdx.x, tid = threadIdx.x;
  const int q = r % slen;
  const int b = (r / slen) / heads;
  float* row = S + (size_t)r * n;
  const unsigned char* mrow = mask + ((size_t)b * slen + q) * (size_t)n;
  float x[4];
  float mx = -1e30f;
#pragma unroll
  for (int i = 0; i < 4; ++i) {
    int c = tid + i * 256;
    float v = row[c];
    if (mrow[c]) v = -1e9f;
    x[i] = v;
    mx = fmaxf(mx, v);
  }
  red[tid] = mx; __syncthreads();
  for (int s = 128; s > 0; s >>= 1) { if (tid < s) red[tid] = fmaxf(red[tid], red[tid + s]); __syncthreads(); }
  mx = red[0]; __syncthreads();
  float sum = 0.0f;
#pragma unroll
  for (int i = 0; i < 4; ++i) { x[i] = __expf(x[i] - mx); sum += x[i]; }
  red[tid] = sum; __syncthreads();
  for (int s = 128; s > 0; s >>= 1) { if (tid < s) red[tid] += red[tid + s]; __syncthreads(); }
  float inv = 1.0f / red[0];
  u16* prow = P + (size_t)r * n;
#pragma unroll
  for (int i = 0; i < 4; ++i) {
    int c = tid + i * 256;
    float v = x[i] * inv;
    if (storeF) row[c] = v;
    prow[c] = f2bf(v);
  }
}

// ----------------- fused bias-residual + LayerNorm -------------------------
// x = X + R ; out = (x-mean)*rsqrt(var+eps)*g + b ; f32 out + optional bf16 out
__global__ __launch_bounds__(256) void bias_res_ln_k(
    const float* __restrict__ X, const float* __restrict__ R,
    const float* __restrict__ g, const float* __restrict__ be,
    float* __restrict__ outF, u16* __restrict__ outB, long ldB, int D) {
  __shared__ float r1[256], r2[256];
  const int row = blockIdx.x, tid = threadIdx.x;
  const float* xr = X + (size_t)row * D;
  const float* rr = R + (size_t)row * D;
  float s = 0.0f, s2 = 0.0f, xv[4];
#pragma unroll
  for (int i = 0; i < 4; ++i) {
    int c = tid + i * 256;
    float v = xr[c] + rr[c];
    xv[i] = v; s += v; s2 += v * v;
  }
  r1[tid] = s; r2[tid] = s2; __syncthreads();
  for (int st = 128; st > 0; st >>= 1) {
    if (tid < st) { r1[tid] += r1[tid + st]; r2[tid] += r2[tid + st]; }
    __syncthreads();
  }
  float mean = r1[0] / D;
  float var  = r2[0] / D - mean * mean;
  float inv  = rsqrtf(var + LN_EPS);
#pragma unroll
  for (int i = 0; i < 4; ++i) {
    int c = tid + i * 256;
    float v = (xv[i] - mean) * inv * g[c] + be[c];
    if (outF) outF[(size_t)row * D + c] = v;
    if (outB) outB[(size_t)row * ldB + c] = f2bf(v);
  }
}

// ---------------------------------------------------------------------------
extern "C" void kernel_launch(void* const* d_in, const int* in_sizes, int n_in,
                              void* d_out, int out_size, void* d_ws, size_t ws_size,
                              hipStream_t stream) {
  (void)in_sizes; (void)n_in; (void)out_size; (void)ws_size;
  const int Bb = 8, S = 1024, D = 1024;
  const int H = 16, HL = 4, DKL = 256, DVL = 256, DKs = 64, DVs = 64;
  const int DFF = 4096;
  const int BS = Bb * S;                 // 8192
  const size_t MD = (size_t)BS * D;      // 8M elements

  const float* enc   = (const float*)d_in[0];
  const unsigned char* mask_l = (const unsigned char*)d_in[1];
  const unsigned char* mask_s = (const unsigned char*)d_in[2];
  const float* wq_l = (const float*)d_in[3];
  const float* wk_l = (const float*)d_in[4];
  const float* wv_l = (const float*)d_in[5];
  const float* wo_l = (const float*)d_in[6];
  const float* bo_l = (const float*)d_in[7];
  const float* g_l  = (const float*)d_in[8];
  const float* b_l  = (const float*)d_in[9];
  const float* wq_s = (const float*)d_in[10];
  const float* wk_s = (const float*)d_in[11];
  const float* wv_s = (const float*)d_in[12];
  const float* wo_s = (const float*)d_in[13];
  const float* bo_s = (const float*)d_in[14];
  const float* g_s  = (const float*)d_in[15];
  const float* b_s  = (const float*)d_in[16];
  const float* w1   = (const float*)d_in[17];
  const float* b1   = (const float*)d_in[18];
  const float* w2   = (const float*)d_in[19];
  const float* b2   = (const float*)d_in[20];
  const float* g_f  = (const float*)d_in[21];
  const float* b_f  = (const float*)d_in[22];

  float* out0 = (float*)d_out;                    // [B,S,D]
  float* attn = out0 + MD;                        // [B,H,S,S] slf_attn

  // ----- workspace carve (256B aligned) -----
  char* p = (char*)d_ws;
  auto alloc = [&](size_t bytes) -> void* {
    void* r = (void*)p; p += (bytes + 255) & ~(size_t)255; return r;
  };
  u16* xb     = (u16*)alloc(MD * 2);
  u16* wqb_l  = (u16*)alloc((size_t)D * 1024 * 2);
  u16* wkb_l  = (u16*)alloc((size_t)D * 1024 * 2);
  u16* wvb_l  = (u16*)alloc((size_t)D * 1024 * 2);
  u16* wob_l  = (u16*)alloc((size_t)1024 * D * 2);
  u16* wqb_s  = (u16*)alloc((size_t)D * 1024 * 2);
  u16* wkb_s  = (u16*)alloc((size_t)D * 1024 * 2);
  u16* wvb_s  = (u16*)alloc((size_t)D * 1024 * 2);
  u16* wob_s  = (u16*)alloc((size_t)1024 * D * 2);
  u16* w1b    = (u16*)alloc((size_t)2 * D * DFF * 2);
  u16* w2b    = (u16*)alloc((size_t)DFF * D * 2);
  u16* ql     = (u16*)alloc(MD * 2);
  u16* kl     = (u16*)alloc(MD * 2);
  u16* vl     = (u16*)alloc(MD * 2);
  float* s_l  = (float*)alloc((size_t)Bb * HL * S * S * 4);
  u16* p_l    = (u16*)alloc((size_t)Bb * HL * S * S * 2);
  u16* ob_l   = (u16*)alloc(MD * 2);
  float* proj_l   = (float*)alloc(MD * 4);
  float* local_f  = (float*)alloc(MD * 4);
  u16* concat = (u16*)alloc((size_t)BS * 2 * D * 2);   // [BS, 2048] bf16
  u16* qs     = (u16*)alloc(MD * 2);
  u16* ks     = (u16*)alloc(MD * 2);
  u16* vs     = (u16*)alloc(MD * 2);
  u16* p_s    = (u16*)alloc((size_t)Bb * H * S * S * 2);
  u16* ob_s   = (u16*)alloc(MD * 2);
  float* proj_s = (float*)alloc(MD * 4);
  float* enc_f  = (float*)alloc(MD * 4);
  u16* h_ff   = (u16*)alloc((size_t)BS * DFF * 2);
  float* y_ff = (float*)alloc(MD * 4);

  auto g1 = [](size_t n) { return dim3((unsigned)((n + 255) / 256)); };

  // ----- precision casts -----
  cast_f32_bf16_k<<<g1(MD), 256, 0, stream>>>(enc, xb, MD);
  cast_head_w_k<<<g1((size_t)HL * D * DKL), 256, 0, stream>>>(wq_l, wqb_l, HL, D, DKL);
  cast_head_w_k<<<g1((size_t)HL * D * DKL), 256, 0, stream>>>(wk_l, wkb_l, HL, D, DKL);
  cast_head_w_k<<<g1((size_t)HL * D * DVL), 256, 0, stream>>>(wv_l, wvb_l, HL, D, DVL);
  cast_f32_bf16_k<<<g1((size_t)1024 * D), 256, 0, stream>>>(wo_l, wob_l, (size_t)1024 * D);
  cast_head_w_k<<<g1((size_t)H * D * DKs), 256, 0, stream>>>(wq_s, wqb_s, H, D, DKs);
  cast_head_w_k<<<g1((size_t)H * D * DKs), 256, 0, stream>>>(wk_s, wkb_s, H, D, DKs);
  cast_head_w_k<<<g1((size_t)H * D * DVs), 256, 0, stream>>>(wv_s, wvb_s, H, D, DVs);
  cast_f32_bf16_k<<<g1((size_t)1024 * D), 256, 0, stream>>>(wo_s, wob_s, (size_t)1024 * D);
  cast_f32_bf16_k<<<g1((size_t)2 * D * DFF), 256, 0, stream>>>(w1, w1b, (size_t)2 * D * DFF);
  cast_f32_bf16_k<<<g1((size_t)DFF * D), 256, 0, stream>>>(w2, w2b, (size_t)DFF * D);

  // ===== local attention =====
  // QKV projections: [8192,1024] x [1024,1024] -> bf16
  gemm_bf16_wmma<128, 128, 4, 2, false, false><<<dim3(8, 64, 1), 256, 0, stream>>>(
      xb, 1024, 0, 0, wqb_l, 1024, 0, 0, nullptr, ql, 1024, 0, 0, nullptr, 1.0f, BS, 1024, 1024, 1);
  gemm_bf16_wmma<128, 128, 4, 2, false, false><<<dim3(8, 64, 1), 256, 0, stream>>>(
      xb, 1024, 0, 0, wkb_l, 1024, 0, 0, nullptr, kl, 1024, 0, 0, nullptr, 1.0f, BS, 1024, 1024, 1);
  gemm_bf16_wmma<128, 128, 4, 2, false, false><<<dim3(8, 64, 1), 256, 0, stream>>>(
      xb, 1024, 0, 0, wvb_l, 1024, 0, 0, nullptr, vl, 1024, 0, 0, nullptr, 1.0f, BS, 1024, 1024, 1);
  // scores = Q Kt / 16 : batched over (b,h)=32
  gemm_bf16_wmma<128, 128, 4, 2, true, false><<<dim3(8, 8, 32), 256, 0, stream>>>(
      ql, 1024, (long)S * 1024, DKL, kl, 1024, (long)S * 1024, DKL,
      s_l, nullptr, S, (long)HL * S * S, (long)S * S, nullptr, 0.0625f, S, S, DKL, HL);
  softmax_mask_k<<<dim3(Bb * HL * S), 256, 0, stream>>>(s_l, mask_l, p_l, S, S, HL, 0);
  // O = P V : batched, N = DVL = 256
  gemm_bf16_wmma<128, 128, 4, 2, false, false><<<dim3(2, 8, 32), 256, 0, stream>>>(
      p_l, 1024, (long)HL * S * S, (long)S * S, vl, 1024, (long)S * 1024, DVL,
      nullptr, ob_l, 1024, (long)S * 1024, DVL, nullptr, 1.0f, S, DVL, S, HL);
  // output projection + bias
  gemm_bf16_wmma<128, 128, 4, 2, false, false><<<dim3(8, 64, 1), 256, 0, stream>>>(
      ob_l, 1024, 0, 0, wob_l, 1024, 0, 0, proj_l, nullptr, 1024, 0, 0, bo_l, 1.0f, BS, D, 1024, 1);
  // residual(enc_input) + LN -> local_f (f32) and concat[:, 0:1024] (bf16)
  bias_res_ln_k<<<dim3(BS), 256, 0, stream>>>(proj_l, enc, g_l, b_l, local_f, concat, 2 * D, D);

  // ===== self attention (on local_out) =====
  gemm_bf16_wmma<128, 128, 4, 2, false, false><<<dim3(8, 64, 1), 256, 0, stream>>>(
      concat, 2 * D, 0, 0, wqb_s, 1024, 0, 0, nullptr, qs, 1024, 0, 0, nullptr, 1.0f, BS, 1024, D, 1);
  gemm_bf16_wmma<128, 128, 4, 2, false, false><<<dim3(8, 64, 1), 256, 0, stream>>>(
      concat, 2 * D, 0, 0, wkb_s, 1024, 0, 0, nullptr, ks, 1024, 0, 0, nullptr, 1.0f, BS, 1024, D, 1);
  gemm_bf16_wmma<128, 128, 4, 2, false, false><<<dim3(8, 64, 1), 256, 0, stream>>>(
      concat, 2 * D, 0, 0, wvb_s, 1024, 0, 0, nullptr, vs, 1024, 0, 0, nullptr, 1.0f, BS, 1024, D, 1);
  // scores -> straight into d_out slf_attn region (f32), batched over 128 (b,h)
  gemm_bf16_wmma<128, 128, 4, 2, true, false><<<dim3(8, 8, 128), 256, 0, stream>>>(
      qs, 1024, (long)S * 1024, DKs, ks, 1024, (long)S * 1024, DKs,
      attn, nullptr, S, (long)H * S * S, (long)S * S, nullptr, 0.125f, S, S, DKs, H);
  // softmax in-place (these probs ARE the slf_attn output) + bf16 copy
  softmax_mask_k<<<dim3(Bb * H * S), 256, 0, stream>>>(attn, mask_s, p_s, S, S, H, 1);
  // O = P V : N = DVs = 64 -> 64x64 tile variant
  gemm_bf16_wmma<64, 64, 2, 2, false, false><<<dim3(1, 16, 128), 128, 0, stream>>>(
      p_s, 1024, (long)H * S * S, (long)S * S, vs, 1024, (long)S * 1024, DVs,
      nullptr, ob_s, 1024, (long)S * 1024, DVs, nullptr, 1.0f, S, DVs, S, H);
  gemm_bf16_wmma<128, 128, 4, 2, false, false><<<dim3(8, 64, 1), 256, 0, stream>>>(
      ob_s, 1024, 0, 0, wob_s, 1024, 0, 0, proj_s, nullptr, 1024, 0, 0, bo_s, 1.0f, BS, D, 1024, 1);
  // residual(local_out) + LN -> enc_f (f32) and concat[:, 1024:2048] (bf16)
  bias_res_ln_k<<<dim3(BS), 256, 0, stream>>>(proj_s, local_f, g_s, b_s, enc_f, concat + D, 2 * D, D);

  // ===== FFN on concat(local_out, enc_out) =====
  gemm_bf16_wmma<128, 128, 4, 2, false, true><<<dim3(32, 64, 1), 256, 0, stream>>>(
      concat, 2 * D, 0, 0, w1b, DFF, 0, 0, nullptr, h_ff, DFF, 0, 0, b1, 1.0f, BS, DFF, 2 * D, 1);
  gemm_bf16_wmma<128, 128, 4, 2, false, false><<<dim3(8, 64, 1), 256, 0, stream>>>(
      h_ff, DFF, 0, 0, w2b, 1024, 0, 0, y_ff, nullptr, 1024, 0, 0, b2, 1.0f, BS, D, DFF, 1);
  // final residual(enc_out) + LN -> d_out[0:B*S*D]
  bias_res_ln_k<<<dim3(BS), 256, 0, stream>>>(y_ff, enc_f, g_f, b_f, out0, nullptr, 0, D);
}
